// FixedSemanticVQGAN_68023692034244
// MI455X (gfx1250) — matrix operand order repaired and checked by reading, not compiled, not run on hardware
//
#include <hip/hip_runtime.h>
#include <hip/hip_bf16.h>
#include <math.h>

// ---------------------------------------------------------------------------
// VQ-VAE quantizer for MI455X (gfx1250, wave32, WMMA + async global->LDS)
//   inputs          [32,256,32,32] f32   -> flat [N=32768, C=256]
//   embeddings      [K=8192, C=256] f32
//   ema_cluster_size[K] f32, ema_embeddings [K,C] f32
// Outputs (concat): quantized_st [32,256,32,32], loss, perplexity,
//                   new_cluster_size[K], new_ema_embeddings[K,C], new_embeddings[K,C]
//
// Hot loop: each wave register-blocks M=32 (two A sets, two accumulators) so
// every B fragment read from LDS feeds two v_wmma_f32_16x16x32_bf16 — halves
// LDS bandwidth per WMMA (the limiter), while E streams through a
// double-buffered ASYNCcnt global->LDS pipeline.
// ---------------------------------------------------------------------------

#define N_ROWS   32768      // 32 * 32 * 32
#define K_CODES  8192
#define C_DIM    256
#define DECAYF   0.99f
#define ONE_M_D  0.01f
#define EPSF     1e-5f

typedef __attribute__((ext_vector_type(16))) __bf16 v16bf;
typedef __attribute__((ext_vector_type(8)))  __bf16 v8bf;
typedef __attribute__((ext_vector_type(8)))  float  v8f;
typedef int v4i __attribute__((vector_size(16)));   // matches builtin param type

#define GLOBAL_AS __attribute__((address_space(1)))
#define LDS_AS    __attribute__((address_space(3)))

// Async global->LDS path (CDNA5 GLOBAL_LOAD_ASYNC_TO_LDS_B128 + ASYNCcnt).
#if defined(__AMDGCN__) && \
    __has_builtin(__builtin_amdgcn_global_load_async_to_lds_b128) && \
    __has_builtin(__builtin_amdgcn_s_wait_asynccnt)
#define HAVE_ASYNC_LDS 1
#else
#define HAVE_ASYNC_LDS 0
#endif

// ---------------- LDS layout for the hot kernel (double buffer) ------------
#define EBUF_BYTES   32768u                 // 64 codes * 256 bf16
#define NORM_OFF     (2u * EBUF_BYTES)      // two E buffers, then two norm bufs
#define SMEM_TOTAL   (2u * EBUF_BYTES + 2u * 64u * sizeof(float))   // 66048 B

// ---------------- workspace layout (bytes) ----------------
#define XB_OFF   0u                          // 32768*256*2 = 16777216
#define EB_OFF   16777216u                   //  8192*256*2 =  4194304
#define EN_OFF   20971520u                   //  8192*4     =    32768
#define IDX_OFF  21004288u                   // 32768*4     =   131072
#define CNT_OFF  21135360u                   //  8192*4     =    32768
#define DW_OFF   21168128u                   //  8192*256*4 =  8388608
#define SC_OFF   29556736u                   //  scalars (8 floats)
#define ZERO_FLOATS ((8192) + (8192*256) + 8)   // counts + dw + scalars (contiguous)

// ---------------- output layout (floats) ----------------
#define OUT_Q     0u
#define OUT_LOSS  8388608u
#define OUT_PERP  8388609u
#define OUT_NCS   8388610u
#define OUT_NEMA  8396802u
#define OUT_NEMB  10493954u

// ===========================================================================
// 0) zero the accumulator region of the workspace (counts, dw, scalars)
// ===========================================================================
__global__ void vq_zero_kernel(float* __restrict__ p, int n) {
    int i = blockIdx.x * blockDim.x + threadIdx.x;
    if (i < n) p[i] = 0.0f;
}

// ===========================================================================
// 1a) inputs [B,C,H,W] f32  ->  x_bf16 [N, C] (N = b*HW + hw), row-major
// ===========================================================================
__global__ void vq_prep_x_kernel(const float* __restrict__ in, __bf16* __restrict__ xb) {
    unsigned i = blockIdx.x * 256u + threadIdx.x;   // i = n*256 + c, total N*C
    unsigned c = i & 255u;
    unsigned n = i >> 8;
    unsigned b = n >> 10;
    unsigned hw = n & 1023u;
    float v = in[(((size_t)b * 256u + c) << 10) + hw];
    xb[i] = (__bf16)v;
}

// ===========================================================================
// 1b) embeddings f32 -> e_bf16 [K,C]  and  eNorm[k] = ||e_k||^2
// ===========================================================================
__global__ void vq_prep_e_kernel(const float* __restrict__ emb,
                                 __bf16* __restrict__ eb,
                                 float* __restrict__ eNorm) {
    __shared__ float red[256];
    int k = blockIdx.x, c = threadIdx.x;
    float v = emb[(size_t)k * C_DIM + c];
    eb[(size_t)k * C_DIM + c] = (__bf16)v;
    red[c] = v * v;
    __syncthreads();
    for (int s = 128; s > 0; s >>= 1) {
        if (c < s) red[c] += red[c + s];
        __syncthreads();
    }
    if (c == 0) eNorm[k] = red[0];
}

// ===========================================================================
// 2) Fused GEMM + argmin:  for each row n, argmin_k ( ||e_k||^2 - 2 x_n.e_k )
//    256 blocks x 4 waves x 32 rows; A fragments resident in VGPRs.
// ===========================================================================
__global__ void __launch_bounds__(128)
vq_argmin_kernel(const __bf16* __restrict__ xb,
                 const __bf16* __restrict__ eb,
                 const float*  __restrict__ eNorm,
                 int* __restrict__ idxOut) {
    extern __shared__ char smem[];

    const int lane   = threadIdx.x & 31;
    const int wave   = threadIdx.x >> 5;                 // 0..3
    const int laneLo = lane & 15;
    const int laneHi = lane >> 4;                        // 0 or 1
    const int rowBase = blockIdx.x * 128 + wave * 32;    // wave owns 32 rows

    // --- load A fragments: two 16-row sets, 8 fragments (16x32 bf16) each.
    // Layout: lanes 0-15 hold K=0..15, lanes 16-31 hold K=16..31 (matches B).
    v16bf afrag0[8], afrag1[8];
    {
        const __bf16* xrow0 = xb + (size_t)(rowBase + laneLo) * C_DIM;
        const __bf16* xrow1 = xb + (size_t)(rowBase + 16 + laneLo) * C_DIM;
        #pragma unroll
        for (int kk = 0; kk < 8; ++kk) {
            const int kb = kk * 32 + (laneHi ? 16 : 0);
            union { v16bf v; v8bf h[2]; } u0, u1;
            u0.h[0] = *(const v8bf*)(xrow0 + kb);
            u0.h[1] = *(const v8bf*)(xrow0 + kb + 8);
            u1.h[0] = *(const v8bf*)(xrow1 + kb);
            u1.h[1] = *(const v8bf*)(xrow1 + kb + 8);
            afrag0[kk] = u0.v;
            afrag1[kk] = u1.v;
        }
    }

    float minD0[8], minD1[8];
    int   minI0[8], minI1[8];
    #pragma unroll
    for (int r = 0; r < 8; ++r) {
        minD0[r] = 3.4e38f; minI0[r] = 0;
        minD1[r] = 3.4e38f; minI1[r] = 0;
    }

    // ---- per-chunk compute: 4 j-tiles of 16 codes; each B fragment read from
    //      LDS feeds TWO WMMAs (M register blocking) ----
    auto compute = [&](const __bf16* eTile, const float* nTile, int codeBase) {
        #pragma unroll
        for (int jt = 0; jt < 4; ++jt) {
            const int jb = jt * 16;
            v8f acc0 = {0.f, 0.f, 0.f, 0.f, 0.f, 0.f, 0.f, 0.f};
            v8f acc1 = {0.f, 0.f, 0.f, 0.f, 0.f, 0.f, 0.f, 0.f};
            // B layout: lane half 0 -> K=kb..kb+15 of column (jb+laneLo),
            //           lane half 1 -> K=kb+16..kb+31 (contiguous 32B per lane)
            const __bf16* ecol = eTile + (jb + laneLo) * C_DIM + (laneHi ? 16 : 0);
            #pragma unroll
            for (int kk = 0; kk < 8; ++kk) {
                v16bf bfrag = *(const v16bf*)(ecol + kk * 32);
                acc0 = __builtin_amdgcn_wmma_f32_16x16x32_bf16(
                           false, afrag0[kk], false, bfrag,
                           (short)0, acc0, false, false);
                acc1 = __builtin_amdgcn_wmma_f32_16x16x32_bf16(
                           false, afrag1[kk], false, bfrag,
                           (short)0, acc1, false, false);
            }
            const int   code = codeBase + jb + laneLo;
            const float en   = nTile[jb + laneLo];
            #pragma unroll
            for (int r = 0; r < 8; ++r) {
                float d0 = en - 2.0f * acc0[r];
                float d1 = en - 2.0f * acc1[r];
                if (d0 < minD0[r]) { minD0[r] = d0; minI0[r] = code; }
                if (d1 < minD1[r]) { minD1[r] = d1; minI1[r] = code; }
            }
        }
    };

#if HAVE_ASYNC_LDS
    // ---- double-buffered ASYNCcnt pipeline: issue next chunk while computing
    auto issue = [&](int chunk, int p) {
        char* dstE = smem + (unsigned)p * EBUF_BYTES;
        const char* srcE = (const char*)(eb + (size_t)chunk * C_DIM);
        // 32 KB chunk = 16 x b128 per thread (128 threads), fully coalesced
        #pragma unroll
        for (int t = 0; t < 16; ++t) {
            __builtin_amdgcn_global_load_async_to_lds_b128(
                (GLOBAL_AS v4i*)(srcE + (unsigned)threadIdx.x * 16u + (unsigned)t * 2048u),
                (LDS_AS v4i*)(dstE + (unsigned)threadIdx.x * 16u + (unsigned)t * 2048u),
                0, 0);
        }
        float* dstN = (float*)(smem + NORM_OFF) + p * 64;
        if (threadIdx.x < 64) dstN[threadIdx.x] = eNorm[chunk + threadIdx.x];
    };

    issue(0, 0);
    for (int c = 0; c < 127; ++c) {
        const int p = c & 1;
        issue(c + 1, p ^ 1);                     // prefetch next chunk (lookahead)
        __builtin_amdgcn_s_wait_asynccnt(16);    // drain current chunk's 16 b128s
        __syncthreads();
        compute((const __bf16*)(smem + (unsigned)p * EBUF_BYTES),
                (const float*)(smem + NORM_OFF) + p * 64, c * 64);
        __syncthreads();                          // protect buffer reuse (WAR)
    }
    {   // tail chunk: nothing left to prefetch
        __builtin_amdgcn_s_wait_asynccnt(0);
        __syncthreads();
        const int p = 127 & 1;
        compute((const __bf16*)(smem + (unsigned)p * EBUF_BYTES),
                (const float*)(smem + NORM_OFF) + p * 64, 127 * 64);
    }
#else
    // ---- synchronous fallback (single buffer, known-good) ----
    {
        __bf16* eTile = (__bf16*)smem;
        float*  nTile = (float*)(smem + NORM_OFF);
        for (int chunk = 0; chunk < K_CODES; chunk += 64) {
            __syncthreads();
            const v8bf* src = (const v8bf*)(eb + (size_t)chunk * C_DIM);
            v8bf* dst = (v8bf*)eTile;
            #pragma unroll
            for (int t = 0; t < 16; ++t)
                dst[threadIdx.x + 128 * t] = src[threadIdx.x + 128 * t];
            if (threadIdx.x < 64) nTile[threadIdx.x] = eNorm[chunk + threadIdx.x];
            if (chunk + 64 < K_CODES)
                __builtin_prefetch(eb + (size_t)(chunk + 64) * C_DIM + threadIdx.x * 128, 0, 0);
            __syncthreads();
            compute(eTile, nTile, chunk);
        }
    }
#endif

    // reduce the 16 column-slots per row (lanes 0-15 = row r, lanes 16-31 = row r+8)
    #pragma unroll
    for (int r = 0; r < 8; ++r) {
        float d = minD0[r];
        int   i = minI0[r];
        #pragma unroll
        for (int m = 8; m >= 1; m >>= 1) {
            float od = __shfl_xor(d, m, 32);
            int   oi = __shfl_xor(i, m, 32);
            if (od < d || (od == d && oi < i)) { d = od; i = oi; }
        }
        if (laneLo == 0) idxOut[rowBase + r + laneHi * 8] = i;
    }
    #pragma unroll
    for (int r = 0; r < 8; ++r) {
        float d = minD1[r];
        int   i = minI1[r];
        #pragma unroll
        for (int m = 8; m >= 1; m >>= 1) {
            float od = __shfl_xor(d, m, 32);
            int   oi = __shfl_xor(i, m, 32);
            if (od < d || (od == d && oi < i)) { d = od; i = oi; }
        }
        if (laneLo == 0) idxOut[rowBase + 16 + r + laneHi * 8] = i;
    }
}

// ===========================================================================
// 3) gather quantized output (NCHW), counts, dw, sum of squared diffs
// ===========================================================================
__global__ void vq_scatter_kernel(const float* __restrict__ in,
                                  const float* __restrict__ emb,
                                  const int*   __restrict__ idx,
                                  float* __restrict__ outq,
                                  float* __restrict__ counts,
                                  float* __restrict__ dw,
                                  float* __restrict__ scalars) {
    __shared__ float red[256];
    const int n = blockIdx.x, c = threadIdx.x;
    const int j = idx[n];
    const int b = n >> 10, hw = n & 1023;
    const size_t ii = (((size_t)b * 256 + c) << 10) + hw;  // NCHW index
    float x = in[ii];
    float q = emb[(size_t)j * C_DIM + c];
    outq[ii] = q;                                  // straight-through == quantized
    atomicAdd(&dw[(size_t)j * C_DIM + c], x);
    float d = q - x;
    red[c] = d * d;
    __syncthreads();
    for (int s = 128; s > 0; s >>= 1) {
        if (c < s) red[c] += red[c + s];
        __syncthreads();
    }
    if (c == 0) {
        atomicAdd(&scalars[0], red[0]);            // sum of squared diffs
        atomicAdd(&counts[j], 1.0f);
    }
}

// ===========================================================================
// 4) global scalars: n_total = sum(decayed sizes), entropy term
// ===========================================================================
__global__ void vq_stats_kernel(const float* __restrict__ counts,
                                const float* __restrict__ ema_cs,
                                float* __restrict__ scalars) {
    int k = blockIdx.x * 256 + threadIdx.x;
    if (k >= K_CODES) return;
    float raw = DECAYF * ema_cs[k] + ONE_M_D * counts[k];
    atomicAdd(&scalars[1], raw);
    float p = counts[k] * (1.0f / (float)N_ROWS);
    atomicAdd(&scalars[2], p * __logf(p + 1e-10f));
}

// ===========================================================================
// 5) finalize: new_cluster_size, new_ema_embeddings, new_embeddings,
//    loss = 1.25 * MSE, perplexity = exp(-entropy)
// ===========================================================================
__global__ void vq_finalize_kernel(const float* __restrict__ counts,
                                   const float* __restrict__ ema_cs,
                                   const float* __restrict__ ema_emb,
                                   const float* __restrict__ dw,
                                   const float* __restrict__ scalars,
                                   float* __restrict__ out) {
    const int k = blockIdx.x, c = threadIdx.x;
    float raw  = DECAYF * ema_cs[k] + ONE_M_D * counts[k];
    float ntot = scalars[1];
    float cs   = (raw + EPSF) / (ntot + (float)K_CODES * EPSF) * ntot;
    float ne   = DECAYF * ema_emb[(size_t)k * C_DIM + c] + ONE_M_D * dw[(size_t)k * C_DIM + c];
    out[OUT_NEMA + (size_t)k * C_DIM + c] = ne;
    out[OUT_NEMB + (size_t)k * C_DIM + c] = ne / (cs + EPSF);
    if (c == 0) {
        out[OUT_NCS + k] = cs;
        if (k == 0) {
            out[OUT_LOSS] = 1.25f * scalars[0] * (1.0f / (float)(N_ROWS * C_DIM));
            out[OUT_PERP] = __expf(-scalars[2]);
        }
    }
}

// ===========================================================================
extern "C" void kernel_launch(void* const* d_in, const int* in_sizes, int n_in,
                              void* d_out, int out_size, void* d_ws, size_t ws_size,
                              hipStream_t stream) {
    const float* inputs   = (const float*)d_in[0];   // [32,256,32,32]
    const float* emb      = (const float*)d_in[1];   // [8192,256]
    const float* ema_cs   = (const float*)d_in[2];   // [8192]
    const float* ema_emb  = (const float*)d_in[3];   // [8192,256]
    float* out = (float*)d_out;
    char*  ws  = (char*)d_ws;

    __bf16* xb     = (__bf16*)(ws + XB_OFF);
    __bf16* ebuf   = (__bf16*)(ws + EB_OFF);
    float*  eNorm  = (float*) (ws + EN_OFF);
    int*    idx    = (int*)   (ws + IDX_OFF);
    float*  counts = (float*) (ws + CNT_OFF);
    float*  dw     = (float*) (ws + DW_OFF);
    float*  sc     = (float*) (ws + SC_OFF);

    // 0) zero accumulators (counts + dw + scalars are contiguous)
    vq_zero_kernel<<<(ZERO_FLOATS + 255) / 256, 256, 0, stream>>>(counts, ZERO_FLOATS);

    // 1) precision/layout prep
    vq_prep_x_kernel<<<(N_ROWS * C_DIM) / 256, 256, 0, stream>>>(inputs, xb);
    vq_prep_e_kernel<<<K_CODES, 256, 0, stream>>>(emb, ebuf, eNorm);

    // 2) fused WMMA distance GEMM + argmin (the 137 GFLOP hot loop)
    //    256 blocks x 128 threads (4 waves x 32 rows = 128 rows/block)
    vq_argmin_kernel<<<N_ROWS / 128, 128, SMEM_TOTAL, stream>>>(xb, ebuf, eNorm, idx);

    // 3) gather + segment sums + loss accumulation
    vq_scatter_kernel<<<N_ROWS, 256, 0, stream>>>(inputs, emb, idx, out + OUT_Q,
                                                  counts, dw, sc);

    // 4) global reductions over K
    vq_stats_kernel<<<(K_CODES + 255) / 256, 256, 0, stream>>>(counts, ema_cs, sc);

    // 5) EMA updates + scalar outputs
    vq_finalize_kernel<<<K_CODES, 256, 0, stream>>>(counts, ema_cs, ema_emb, dw, sc, out);
}
